// DecoderModel_53343493816711
// MI455X (gfx1250) — matrix operand order repaired and checked by reading, not compiled, not run on hardware
//
#include <hip/hip_runtime.h>
#include <hip/hip_bf16.h>
#include <stdint.h>

// ---------------- model constants ----------------
#define S_LEN   2048
#define H_DIM   2048
#define N_HEADS 32
#define KV_DIM  512          // KV * HD = 8*64
#define HEAD_D  64
#define FF_I    8192
#define N_LAYER 2
#define NEGF    (-65504.0f)  // float16 min, per reference
#define INV_SCALE 0.125f     // 1/sqrt(64)

typedef __bf16 bf16;
typedef __attribute__((ext_vector_type(16))) __bf16 v16bf;
typedef __attribute__((ext_vector_type(8)))  float  v8f;
typedef unsigned int u32x4 __attribute__((ext_vector_type(4)));
typedef unsigned int u32x8 __attribute__((ext_vector_type(8)));

union BFrag {
    v16bf        v;
    unsigned int u[8];
    uint4        u4[2];
};

__device__ __forceinline__ unsigned short bfbits(float f) {
    bf16 h = (bf16)f;
    return __builtin_bit_cast(unsigned short, h);
}
__device__ __forceinline__ unsigned int pack2(float lo, float hi) {
    return (unsigned int)bfbits(lo) | ((unsigned int)bfbits(hi) << 16);
}

// ---------------- RMSNorm: f32 in -> bf16 out (row = 2048) ----------------
__global__ __launch_bounds__(256)
void rmsnorm_kernel(const float* __restrict__ x, const float* __restrict__ w,
                    bf16* __restrict__ out)
{
    __shared__ float red[8];
    const int row = blockIdx.x;
    const float* xr = x + (size_t)row * H_DIM;
    const int base = threadIdx.x * 8;
    float4 a = *(const float4*)(xr + base);
    float4 b = *(const float4*)(xr + base + 4);
    float ss = a.x*a.x + a.y*a.y + a.z*a.z + a.w*a.w
             + b.x*b.x + b.y*b.y + b.z*b.z + b.w*b.w;
    #pragma unroll
    for (int off = 16; off > 0; off >>= 1) ss += __shfl_xor(ss, off, 32);
    if ((threadIdx.x & 31) == 0) red[threadIdx.x >> 5] = ss;
    __syncthreads();
    float tot = red[0]+red[1]+red[2]+red[3]+red[4]+red[5]+red[6]+red[7];
    const float rs = rsqrtf(tot * (1.0f / H_DIM) + 1e-5f);
    const float* wr = w + base;
    bf16* o = out + (size_t)row * H_DIM + base;
    float v[8] = {a.x,a.y,a.z,a.w,b.x,b.y,b.z,b.w};
    #pragma unroll
    for (int i = 0; i < 8; ++i) o[i] = (bf16)(wr[i] * v[i] * rs);
}

// ---------------- RoPE: f32 in -> bf16 out ----------------
__global__ __launch_bounds__(256)
void rope_kernel(const float* __restrict__ in, const float* __restrict__ cosT,
                 const float* __restrict__ sinT, const int* __restrict__ pos,
                 bf16* __restrict__ out, int width)
{
    const int idx = blockIdx.x * 256 + threadIdx.x;
    const int s = idx / width;
    const int c = idx - s * width;
    const int d = c & (HEAD_D - 1);
    const int p = pos[s];
    const float v   = in[idx];
    const float rot = (d < HEAD_D/2) ? -in[idx + HEAD_D/2] : in[idx - HEAD_D/2];
    out[idx] = (bf16)(v * cosT[p * HEAD_D + d] + rot * sinT[p * HEAD_D + d]);
}

// ---------------- bf16 WMMA GEMM: C[M,N] = A[M,K](bf16) x B[K,N](f32->bf16) ----
// A tile (128x32 bf16, row-major) is DMA'd global->LDS by the Tensor Data Mover
// (issued once per block by wave 0; TENSORcnt-tracked). B tile is converted
// f32->bf16 while being staged in WMMA B-fragment order so fragment reads are
// two contiguous ds_load_b128 each.
// MODE 0: store f32   MODE 1: store bf16
// MODE 2: store f32 + Res (residual add)
// MODE 3: store bf16 = silu(Res) * acc  (fused SwiGLU epilogue)
template<int MODE>
__global__ __launch_bounds__(256)
void gemm_kernel(const bf16* __restrict__ A, const float* __restrict__ Bw,
                 void* __restrict__ Cout, const float* __restrict__ Res,
                 int M, int N, int K)
{
    __shared__ bf16 As[128 * 32];          // row-major (TDM fills linearly)
    __shared__ bf16 Bs[8 * 2 * 16 * 16];   // [nTile][kHalf][lane][16] fragment order
    const int tid  = threadIdx.x;
    const int wave = tid >> 5;
    const int lane = tid & 31;
    const int bm = blockIdx.y * 128;
    const int bn = blockIdx.x * 128;
    const int m0 = wave * 16;
    const int lm = lane & 15;
    const int kh = lane >> 4;

    // B staging indices: thread packs a 2(k) x 8(n) patch
    const int k0 = (tid >> 4) << 1;        // even k row 0..30
    const int pc = tid & 15;
    const int cc = pc << 3;                // col 0..120
    const int t2 = pc >> 1;                // n-tile
    const int nb = (pc & 1) << 3;          // lane-base within n-tile

    v8f acc[8];
    #pragma unroll
    for (int t = 0; t < 8; ++t) acc[t] = (v8f){0.f,0.f,0.f,0.f,0.f,0.f,0.f,0.f};

    for (int kb = 0; kb < K; kb += 32) {
        // ---- A tile via Tensor Data Mover (wave 0 issues one descriptor) ----
        if (wave == 0) {
            const unsigned long long ga =
                (unsigned long long)(uintptr_t)(A + (size_t)bm * K + kb);
            const unsigned int lds = (unsigned int)(uintptr_t)(&As[0]);
            u32x4 g0;
            g0[0] = 1u;                                   // count=1, user mode
            g0[1] = lds;                                  // lds_addr
            g0[2] = (unsigned int)ga;                     // global_addr[31:0]
            g0[3] = (unsigned int)(ga >> 32) | (2u << 30);// global_addr hi | type=2
            u32x8 g1;
            g1[0] = 1u << 16;                             // data_size=2B, no flags
            g1[1] = ((unsigned int)K & 0xffffu) << 16;    // tensor_dim0 lo16
            g1[2] = ((unsigned int)K >> 16) | (128u << 16); // dim0 hi | tensor_dim1 lo
            g1[3] = 32u << 16;                            // dim1 hi=0 | tile_dim0=32
            g1[4] = 128u;                                 // tile_dim1=128, tile_dim2=0
            g1[5] = (unsigned int)K;                      // tensor_dim0_stride lo32
            g1[6] = 0u;                                   // stride hi | dim1_stride lo
            g1[7] = 0u;
            asm volatile("tensor_load_to_lds %0, %1" :: "s"(g0), "s"(g1) : "memory");
        }
        // ---- B tile: f32 -> bf16, stored in fragment order ----
        {
            const float* r0 = Bw + (size_t)(kb + k0) * N + bn + cc;
            const float* r1 = r0 + N;
            float4 a0 = ((const float4*)r0)[0], a1 = ((const float4*)r0)[1];
            float4 b0 = ((const float4*)r1)[0], b1 = ((const float4*)r1)[1];
            float u0[8] = {a0.x,a0.y,a0.z,a0.w,a1.x,a1.y,a1.z,a1.w};
            float u1[8] = {b0.x,b0.y,b0.z,b0.w,b1.x,b1.y,b1.z,b1.w};
            bf16* dstB = Bs + ((((t2 << 1) | (k0 >> 4)) * 16 + nb) << 4) + (k0 & 15);
            #pragma unroll
            for (int i = 0; i < 8; ++i)
                *(unsigned int*)(dstB + i * 16) = pack2(u0[i], u1[i]);
        }
        __builtin_amdgcn_s_wait_tensorcnt(0);  // no-op for waves 1..7
        __syncthreads();
        if (kb + 32 < K) { // prefetch next B tile -> global_prefetch_b8
            __builtin_prefetch(Bw + (size_t)(kb + 32 + k0) * N + bn + cc, 0, 0);
        }
        BFrag a; // A frag: 16x32, contiguous k = 16*kh + s order gathered from row-major LDS
        #pragma unroll
        for (int j = 0; j < 8; ++j) {
            const int kloc = (kh << 4) | (j << 1);
            a.u[j] = *(const unsigned int*)(As + (m0 + lm) * 32 + kloc);
        }
        #pragma unroll
        for (int t = 0; t < 8; ++t) {
            BFrag b; // already in fragment order: 2x ds_load_b128
            const bf16* bp = Bs + ((((t << 1) | kh) * 16 + lm) << 4);
            b.u4[0] = ((const uint4*)bp)[0];
            b.u4[1] = ((const uint4*)bp)[1];
            acc[t] = __builtin_amdgcn_wmma_f32_16x16x32_bf16(
                         false, a.v, false, b.v, (short)0, acc[t], false, false);
        }
        __syncthreads();
    }

    #pragma unroll
    for (int t = 0; t < 8; ++t) {
        #pragma unroll
        for (int r = 0; r < 8; ++r) {
            const int row = bm + m0 + r + (kh << 3);
            const int col = bn + t * 16 + lm;
            const size_t idx = (size_t)row * N + col;
            const float v = acc[t][r];
            if (MODE == 0)      ((float*)Cout)[idx] = v;
            else if (MODE == 1) ((bf16*)Cout)[idx] = (bf16)v;
            else if (MODE == 2) ((float*)Cout)[idx] = v + Res[idx];
            else {
                const float g = Res[idx];
                const float s = g / (1.f + __expf(-g));
                ((bf16*)Cout)[idx] = (bf16)(s * v);
            }
        }
    }
}

// ---------------- Flash attention (causal, GQA 4:1), bf16 WMMA ----------------
// One wave handles one 16-query tile of one head. Per-wave LDS staging of P
// (C-layout -> A-layout transpose) guarded by s_wait_dscnt (wave-local).
__global__ __launch_bounds__(256)
void attn_kernel(const bf16* __restrict__ Q, const bf16* __restrict__ Kb,
                 const bf16* __restrict__ Vb, const float* __restrict__ amask,
                 bf16* __restrict__ Out)
{
    __shared__ bf16 pstage[8][16 * 32];
    const int lane = threadIdx.x & 31;
    const int wave = threadIdx.x >> 5;
    const int qt   = blockIdx.x * 8 + wave;   // 0..127
    const int head = blockIdx.y;              // 0..31
    const int kvh  = head >> 2;               // GQA: N_REP=4
    const int lm = lane & 15;
    const int kh = lane >> 4;

    BFrag aq[2]; // Q A-frags, d-steps of 32
    {
        const int srow = qt * 16 + lm;
        const bf16* qrow = Q + (size_t)srow * H_DIM + head * HEAD_D;
        #pragma unroll
        for (int st = 0; st < 2; ++st)
            #pragma unroll
            for (int j = 0; j < 8; ++j) {
                const int kloc = st * 32 + ((kh << 4) | (j << 1));
                aq[st].u[j] = *(const unsigned int*)(qrow + kloc);
            }
    }

    v8f o[4];
    #pragma unroll
    for (int t = 0; t < 4; ++t) o[t] = (v8f){0.f,0.f,0.f,0.f,0.f,0.f,0.f,0.f};
    float mi[8], li[8];
    #pragma unroll
    for (int r = 0; r < 8; ++r) { mi[r] = -1e30f; li[r] = 0.f; }

    const int nkc = (qt * 16 + 16 + 31) >> 5; // 32-key chunks (causal bound)
    for (int kc = 0; kc < nkc; ++kc) {
        v8f c[2];
        c[0] = (v8f){0.f,0.f,0.f,0.f,0.f,0.f,0.f,0.f};
        c[1] = (v8f){0.f,0.f,0.f,0.f,0.f,0.f,0.f,0.f};
        #pragma unroll
        for (int s2 = 0; s2 < 2; ++s2) {
            const int key = kc * 32 + s2 * 16 + lm;
            #pragma unroll
            for (int st = 0; st < 2; ++st) {
                BFrag b; // K^T frag: contiguous 16 bf16 of this key's d-slice
                const bf16* krow = Kb + (size_t)key * KV_DIM + kvh * HEAD_D
                                 + st * 32 + (kh << 4);
                b.u4[0] = ((const uint4*)krow)[0];
                b.u4[1] = ((const uint4*)krow)[1];
                c[s2] = __builtin_amdgcn_wmma_f32_16x16x32_bf16(
                            false, aq[st].v, false, b.v, (short)0, c[s2], false, false);
            }
        }
        // scale + causal + pad mask
        float sc[2][8];
        #pragma unroll
        for (int s2 = 0; s2 < 2; ++s2) {
            const int kj = kc * 32 + s2 * 16 + lm;
            const float padd = (amask[kj] != 0.f) ? 0.f : NEGF;
            #pragma unroll
            for (int r = 0; r < 8; ++r) {
                const int qi = qt * 16 + r + (kh << 3);
                sc[s2][r] = c[s2][r] * INV_SCALE + padd + ((kj > qi) ? NEGF : 0.f);
            }
        }
        // online softmax
        float rm[8], alpha[8], rs[8];
        #pragma unroll
        for (int r = 0; r < 8; ++r) {
            rm[r] = fmaxf(sc[0][r], sc[1][r]);
            #pragma unroll
            for (int off = 1; off < 16; off <<= 1)
                rm[r] = fmaxf(rm[r], __shfl_xor(rm[r], off, 16));
            const float mn = fmaxf(mi[r], rm[r]);
            alpha[r] = __expf(mi[r] - mn);
            mi[r] = mn;
        }
        #pragma unroll
        for (int s2 = 0; s2 < 2; ++s2)
            #pragma unroll
            for (int r = 0; r < 8; ++r) {
                const float p = __expf(sc[s2][r] - mi[r]);
                sc[s2][r] = p;
                pstage[wave][(r + (kh << 3)) * 32 + s2 * 16 + lm] = (bf16)p;
            }
        #pragma unroll
        for (int r = 0; r < 8; ++r) {
            rs[r] = sc[0][r] + sc[1][r];
            #pragma unroll
            for (int off = 1; off < 16; off <<= 1)
                rs[r] += __shfl_xor(rs[r], off, 16);
            li[r] = li[r] * alpha[r] + rs[r];
        }
        // wave-local LDS RAW: wait DS ops (no block barrier needed)
        asm volatile("s_wait_dscnt 0x0" ::: "memory");
        BFrag pa; // P as A-matrix (16x32)
        #pragma unroll
        for (int j = 0; j < 8; ++j) {
            const int kloc = (kh << 4) | (j << 1);
            pa.u[j] = *(const unsigned int*)(&pstage[wave][lm * 32 + kloc]);
        }
        #pragma unroll
        for (int t = 0; t < 4; ++t)
            #pragma unroll
            for (int r = 0; r < 8; ++r) o[t][r] *= alpha[r];
        #pragma unroll
        for (int t = 0; t < 4; ++t) {
            BFrag bv; // V frag 32x16 (key x d)
            #pragma unroll
            for (int j = 0; j < 8; ++j) {
                const int key0 = kc * 32 + (kh << 4) + (j << 1);
                const size_t base = (size_t)key0 * KV_DIM + kvh * HEAD_D + t * 16 + lm;
                unsigned int lo = *(const unsigned short*)(Vb + base);
                unsigned int hi = *(const unsigned short*)(Vb + base + KV_DIM);
                bv.u[j] = lo | (hi << 16);
            }
            o[t] = __builtin_amdgcn_wmma_f32_16x16x32_bf16(
                       false, pa.v, false, bv.v, (short)0, o[t], false, false);
        }
    }
    #pragma unroll
    for (int r = 0; r < 8; ++r) {
        const float inv = 1.f / li[r];
        const int srow = qt * 16 + r + (kh << 3);
        #pragma unroll
        for (int t = 0; t < 4; ++t)
            Out[(size_t)srow * H_DIM + head * HEAD_D + t * 16 + lm] = (bf16)(o[t][r] * inv);
    }
}

// ---------------- host orchestration ----------------
extern "C" void kernel_launch(void* const* d_in, const int* in_sizes, int n_in,
                              void* d_out, int out_size, void* d_ws, size_t ws_size,
                              hipStream_t stream) {
    const float* x     = (const float*)d_in[0];
    const float* amask = (const float*)d_in[1];
    const float* cosT  = (const float*)d_in[2];
    const float* sinT  = (const float*)d_in[3];
    const float* Wq    = (const float*)d_in[4];
    const float* Wk    = (const float*)d_in[5];
    const float* Wv    = (const float*)d_in[6];
    const float* Wo    = (const float*)d_in[7];
    const float* Wg    = (const float*)d_in[8];
    const float* Wu    = (const float*)d_in[9];
    const float* Wd    = (const float*)d_in[10];
    const float* ln1   = (const float*)d_in[11];
    const float* ln2   = (const float*)d_in[12];
    const int*   pos   = (const int*)d_in[13];
    float* xO = (float*)d_out;

    char* w = (char*)d_ws;
    auto alloc = [&](size_t bytes) -> char* {
        char* p = w; w += (bytes + 255) & ~(size_t)255; return p;
    };
    bf16*  h_bf = (bf16*) alloc((size_t)S_LEN * H_DIM * 2);
    float* q_f  = (float*)alloc((size_t)S_LEN * H_DIM * 4);
    float* k_f  = (float*)alloc((size_t)S_LEN * KV_DIM * 4);
    bf16*  v_bf = (bf16*) alloc((size_t)S_LEN * KV_DIM * 2);
    bf16*  qr   = (bf16*) alloc((size_t)S_LEN * H_DIM * 2);
    bf16*  kr   = (bf16*) alloc((size_t)S_LEN * KV_DIM * 2);
    bf16*  at   = (bf16*) alloc((size_t)S_LEN * H_DIM * 2);
    float* g_f  = (float*)alloc((size_t)S_LEN * FF_I * 4);
    bf16*  t_bf = (bf16*) alloc((size_t)S_LEN * FF_I * 2);

    // x -> output buffer (residual stream lives in d_out)
    hipMemcpyAsync(xO, x, (size_t)S_LEN * H_DIM * sizeof(float),
                   hipMemcpyDeviceToDevice, stream);

    const dim3 blk(256);
    for (int l = 0; l < N_LAYER; ++l) {
        const float* Wq_l = Wq + (size_t)l * H_DIM * H_DIM;
        const float* Wk_l = Wk + (size_t)l * H_DIM * KV_DIM;
        const float* Wv_l = Wv + (size_t)l * H_DIM * KV_DIM;
        const float* Wo_l = Wo + (size_t)l * H_DIM * H_DIM;
        const float* Wg_l = Wg + (size_t)l * H_DIM * FF_I;
        const float* Wu_l = Wu + (size_t)l * H_DIM * FF_I;
        const float* Wd_l = Wd + (size_t)l * FF_I * H_DIM;

        // 1) h = rmsnorm(x, ln1)  (bf16)
        rmsnorm_kernel<<<S_LEN, blk, 0, stream>>>(xO, ln1 + (size_t)l * H_DIM, h_bf);
        // 2) q/k/v projections
        gemm_kernel<0><<<dim3(H_DIM/128,  S_LEN/128), blk, 0, stream>>>(h_bf, Wq_l, q_f,  nullptr, S_LEN, H_DIM,  H_DIM);
        gemm_kernel<0><<<dim3(KV_DIM/128, S_LEN/128), blk, 0, stream>>>(h_bf, Wk_l, k_f,  nullptr, S_LEN, KV_DIM, H_DIM);
        gemm_kernel<1><<<dim3(KV_DIM/128, S_LEN/128), blk, 0, stream>>>(h_bf, Wv_l, v_bf, nullptr, S_LEN, KV_DIM, H_DIM);
        // 3) RoPE
        rope_kernel<<<(S_LEN * H_DIM) / 256,  blk, 0, stream>>>(q_f, cosT, sinT, pos, qr, H_DIM);
        rope_kernel<<<(S_LEN * KV_DIM) / 256, blk, 0, stream>>>(k_f, cosT, sinT, pos, kr, KV_DIM);
        // 4) flash attention
        attn_kernel<<<dim3(S_LEN/(16*8), N_HEADS), blk, 0, stream>>>(qr, kr, v_bf, amask, at);
        // 5) x += attn @ Wo
        gemm_kernel<2><<<dim3(H_DIM/128, S_LEN/128), blk, 0, stream>>>(at, Wo_l, xO, xO, S_LEN, H_DIM, H_DIM);
        // 6) h2 = rmsnorm(x, ln2)
        rmsnorm_kernel<<<S_LEN, blk, 0, stream>>>(xO, ln2 + (size_t)l * H_DIM, h_bf);
        // 7) g = h2 @ Wg ; t = silu(g) * (h2 @ Wu)   (fused epilogue)
        gemm_kernel<0><<<dim3(FF_I/128, S_LEN/128), blk, 0, stream>>>(h_bf, Wg_l, g_f,  nullptr, S_LEN, FF_I, H_DIM);
        gemm_kernel<3><<<dim3(FF_I/128, S_LEN/128), blk, 0, stream>>>(h_bf, Wu_l, t_bf, g_f,     S_LEN, FF_I, H_DIM);
        // 8) x += t @ Wd
        gemm_kernel<2><<<dim3(H_DIM/128, S_LEN/128), blk, 0, stream>>>(t_bf, Wd_l, xO, xO, S_LEN, H_DIM, FF_I);
    }
    (void)in_sizes; (void)n_in; (void)out_size; (void)ws_size;
}
